// diffAttn_56169582297721
// MI455X (gfx1250) — compile-verified, matrix operand order checked by reading, not run on hardware
//
#include <hip/hip_runtime.h>
#include <hip/hip_bf16.h>
#include <math.h>

// ---------------------------------------------------------------------------
// Types / helpers for CDNA5 WMMA (wave32, 16x16x32 bf16 -> f32)
// ---------------------------------------------------------------------------
typedef __attribute__((ext_vector_type(16))) __bf16 bf16x16;
typedef __attribute__((ext_vector_type(8)))  float  f32x8;

struct alignas(16) F4f { float x, y, z, w; };
struct F8f { F4f lo, hi; };

#define LAMBDA_INIT 0.3555090675909693f   // 0.8 - 0.6*exp(-0.3)

__device__ __forceinline__ f32x8 zero8() {
  f32x8 z;
#pragma unroll
  for (int i = 0; i < 8; ++i) z[i] = 0.0f;
  return z;
}

__device__ __forceinline__ f32x8 wmma_bf16f32(bf16x16 a, bf16x16 b, f32x8 c) {
  // D = A*B + C ; f32 accumulate
  return __builtin_amdgcn_wmma_f32_16x16x32_bf16(false, a, false, b, (short)0, c,
                                                 false, false);
}

// Load a 16x32 bf16 A/B fragment from a row-major [rows][ld] bf16 matrix.
// Per ISA layout: lane L owns row (row0 + L%16); lanes 0-15 hold K 0-7 & 16-23,
// lanes 16-31 hold K 8-15 & 24-31 (2 bf16 per VGPR).
__device__ __forceinline__ bf16x16 load_frag(const __bf16* __restrict__ base, int ld,
                                             int row0, int k0, int lane) {
  int r = row0 + (lane & 15);
  int khalf = (lane >> 4) * 8;
  const __bf16* p = base + (size_t)r * (size_t)ld + k0 + khalf;
  F8f f;
  f.lo = *reinterpret_cast<const F4f*>(p);
  f.hi = *reinterpret_cast<const F4f*>(p + 16);
  return __builtin_bit_cast(bf16x16, f);
}

// ---------------------------------------------------------------------------
// Small setup kernels: lambda scalar, weight transpose+bf16 convert
// ---------------------------------------------------------------------------
__global__ __launch_bounds__(256) void da_lam_kernel(
    const float* __restrict__ lq1, const float* __restrict__ lk1,
    const float* __restrict__ lq2, const float* __restrict__ lk2,
    float* __restrict__ lam_out) {
  __shared__ float r1[256], r2[256];
  int t = threadIdx.x;
  r1[t] = lq1[t] * lk1[t];
  r2[t] = lq2[t] * lk2[t];
  __syncthreads();
  for (int s = 128; s > 0; s >>= 1) {
    if (t < s) { r1[t] += r1[t + s]; r2[t] += r2[t + s]; }
    __syncthreads();
  }
  if (t == 0) lam_out[0] = __expf(r1[0]) - __expf(r2[0]);
}

// WT[n*K + k] = (bf16) W[k*N + n]   (W is [K][N] row-major f32)
__global__ __launch_bounds__(256) void da_tconv_kernel(
    const float* __restrict__ W, __bf16* __restrict__ WT, int K, int N) {
  int idx = blockIdx.x * blockDim.x + threadIdx.x;
  if (idx < K * N) {
    int n = idx / K, k = idx - n * K;
    WT[idx] = (__bf16)W[k * N + n];
  }
}

// ---------------------------------------------------------------------------
// K1: per-batch ln_attn over seq dim + trend/seasonal decomposition.
// Outputs transposed bf16 [B][c][s] so tokens (=channels) are rows.
// Dynamic LDS: 256*257 f32 (padded stride -> conflict-free transpose).
// ---------------------------------------------------------------------------
__global__ __launch_bounds__(256) void da_prep_attn_kernel(
    const float* __restrict__ x, const float* __restrict__ lnscale,
    __bf16* __restrict__ aT, __bf16* __restrict__ trT, __bf16* __restrict__ seT) {
  extern __shared__ float lds[];  // 256*257 floats
  const int SW = 257;
  int t = threadIdx.x, b = blockIdx.x;
  const float* xb = x + (size_t)b * 65536;

  // coalesced load of the whole [256][256] slab (float4 per thread)
  int c4 = (t & 63) * 4, sr = t >> 6;
  for (int s0 = 0; s0 < 256; s0 += 4) {
    int s = s0 + sr;
    F4f v = *reinterpret_cast<const F4f*>(xb + (size_t)s * 256 + c4);
    lds[s * SW + c4 + 0] = v.x; lds[s * SW + c4 + 1] = v.y;
    lds[s * SW + c4 + 2] = v.z; lds[s * SW + c4 + 3] = v.w;
  }
  __syncthreads();

  // column (seq) norm for channel c = t ; banks (s+t)%64 -> conflict-free
  float ss = 0.f;
  for (int s = 0; s < 256; ++s) { float v = lds[s * SW + t]; ss += v * v; }
  float inv = 1.f / (sqrtf(ss) * 0.0625f + 1e-8f);  // rms = ||.||/16
  for (int s = 0; s < 256; ++s) lds[s * SW + t] *= lnscale[s] * inv;
  __syncthreads();

  // write aT/trendT/seasonalT rows (c) with thread = s (coalesced bf16 stores)
  size_t base = (size_t)b * 65536;
  int sm2 = t >= 2 ? t - 2 : 0, sm1 = t >= 1 ? t - 1 : 0;
  int sp1 = t <= 254 ? t + 1 : 255, sp2 = t <= 253 ? t + 2 : 255;
  for (int c = 0; c < 256; ++c) {
    float ac = lds[t * SW + c];
    float tr = (lds[sm2 * SW + c] + lds[sm1 * SW + c] + ac +
                lds[sp1 * SW + c] + lds[sp2 * SW + c]) * 0.2f;
    size_t o = base + (size_t)c * 256 + t;
    aT[o] = (__bf16)ac;
    trT[o] = (__bf16)tr;
    seT[o] = (__bf16)(ac - tr);
  }
}

// ---------------------------------------------------------------------------
// Generic batched bf16 WMMA GEMM:  C[b] = act(A[b] * BT[b]^T + bias)
// A: [M][K] bf16 row-major, BT: [N][K] bf16 row-major (i.e. B transposed).
// Block = 8 waves, tile 64(M) x 256(N); wave tile 32x64 (2 A-frags x 4 B-frags
// -> 8 WMMAs per 6 fragment loads per k-step).
// MODE 0: bf16 row-major store    out[m*ldc+n]
// MODE 1: bf16 transposed store   out[n*ldc+m]
// MODE 2: f32 transposed store with residual add: out[n*ldc+m]=v+res[n*ldc+m]
// ACT  0: none   1: SiLU
// ---------------------------------------------------------------------------
template <int MODE, int ACT>
__global__ __launch_bounds__(256) void da_gemm_kernel(
    const __bf16* __restrict__ A, long long sA, int lda,
    const __bf16* __restrict__ BT, long long sB, int ldb,
    const float* __restrict__ bias,
    void* __restrict__ Cout, long long sC, int ldc,
    const float* __restrict__ resid, long long sR, int K) {
  int lane = threadIdx.x & 31, w = threadIdx.x >> 5;
  int wm = w & 1, wn = w >> 1;
  int b = blockIdx.z;
  int m0 = blockIdx.y * 64 + wm * 32;
  int n0 = blockIdx.x * 256 + wn * 64;
  const __bf16* Ab = A + (size_t)b * sA;
  const __bf16* Bb = BT + (size_t)b * sB;

  f32x8 acc[2][4];
#pragma unroll
  for (int i = 0; i < 2; ++i)
#pragma unroll
    for (int j = 0; j < 4; ++j) acc[i][j] = zero8();

  for (int k0 = 0; k0 < K; k0 += 32) {
    bf16x16 a0 = load_frag(Ab, lda, m0,      k0, lane);
    bf16x16 a1 = load_frag(Ab, lda, m0 + 16, k0, lane);
#pragma unroll
    for (int j = 0; j < 4; ++j) {
      bf16x16 bf = load_frag(Bb, ldb, n0 + 16 * j, k0, lane);
      acc[0][j] = wmma_bf16f32(a0, bf, acc[0][j]);
      acc[1][j] = wmma_bf16f32(a1, bf, acc[1][j]);
    }
  }

#pragma unroll
  for (int i = 0; i < 2; ++i) {
#pragma unroll
    for (int j = 0; j < 4; ++j) {
      int n = n0 + 16 * j + (lane & 15);
      int mb = m0 + 16 * i + 8 * (lane >> 4);
      float bs = bias ? bias[n] : 0.f;
#pragma unroll
      for (int r = 0; r < 8; ++r) {
        int m = mb + r;
        float v = acc[i][j][r] + bs;
        if (ACT == 1) v = v / (1.f + __expf(-v));  // SiLU
        if (MODE == 0) {
          ((__bf16*)Cout)[(size_t)b * sC + (size_t)m * ldc + n] = (__bf16)v;
        } else if (MODE == 1) {
          ((__bf16*)Cout)[(size_t)b * sC + (size_t)n * ldc + m] = (__bf16)v;
        } else {
          size_t o = (size_t)b * sC + (size_t)n * ldc + m;
          ((float*)Cout)[o] = v + resid[(size_t)b * sR + (size_t)n * ldc + m];
        }
      }
    }
  }
}

// ---------------------------------------------------------------------------
// K5: fused differential attention per (64-row block, batch).
// S1,S2 -> softmax -> P = sm1 - lam*sm2 (bf16 LDS) -> P @ V (WMMA) ->
// row rmsnorm * (1-LI) -> x1 = x + attn^T.
// Wave tile 32x64; 8 waves cover the full 64x256 tile in one pass.
// ---------------------------------------------------------------------------
__global__ __launch_bounds__(256) void da_attn_kernel(
    const __bf16* __restrict__ Q, const __bf16* __restrict__ Kmat,
    const __bf16* __restrict__ VT, const float* __restrict__ lamPtr,
    const float* __restrict__ normScale, const float* __restrict__ x,
    float* __restrict__ x1) {
  extern __shared__ char smem[];
  const int SW = 265;  // f32 stride (265%64=9 -> conflict-free transpose reads)
  const int SP = 264;  // bf16 stride (multiple of 8 -> 16B frag alignment)
  float*  s1 = (float*)smem;           // 64*265 f32
  float*  s2 = s1 + 64 * SW;           // 64*265 f32
  __bf16* sp = (__bf16*)(s2 + 64 * SW);// 64*264 bf16
  float*  rowsc = (float*)(sp + 64 * SP); // 64 f32

  int t = threadIdx.x, lane = t & 31, w = t >> 5;
  int wm = w & 1, wn = w >> 1;
  int b = blockIdx.y, mb = blockIdx.x * 64;
  float lam = lamPtr[0];

  const __bf16* Qb = Q + (size_t)b * 131072;
  const __bf16* Kb = Kmat + (size_t)b * 131072;

  // ---- phase 1: S1 (half 0) and S2 (half 1), scaled by E^-0.25 = 0.25 ----
  for (int half = 0; half < 2; ++half) {
    float* sb = half ? s2 : s1;
    const __bf16* Aq = Qb + half * 256;
    const __bf16* Bk = Kb + half * 256;
    int m0 = mb + wm * 32;
    int n0 = wn * 64;
    f32x8 acc[2][4];
#pragma unroll
    for (int i = 0; i < 2; ++i)
#pragma unroll
      for (int j = 0; j < 4; ++j) acc[i][j] = zero8();
    for (int k0 = 0; k0 < 256; k0 += 32) {
      bf16x16 a0 = load_frag(Aq, 512, m0,      k0, lane);
      bf16x16 a1 = load_frag(Aq, 512, m0 + 16, k0, lane);
#pragma unroll
      for (int j = 0; j < 4; ++j) {
        bf16x16 bf = load_frag(Bk, 512, n0 + 16 * j, k0, lane);
        acc[0][j] = wmma_bf16f32(a0, bf, acc[0][j]);
        acc[1][j] = wmma_bf16f32(a1, bf, acc[1][j]);
      }
    }
#pragma unroll
    for (int i = 0; i < 2; ++i)
#pragma unroll
      for (int j = 0; j < 4; ++j)
#pragma unroll
        for (int r = 0; r < 8; ++r) {
          int ml = wm * 32 + 16 * i + 8 * (lane >> 4) + r;
          int nl = n0 + 16 * j + (lane & 15);
          sb[ml * SW + nl] = acc[i][j][r] * 0.25f;
        }
  }
  __syncthreads();

  // ---- phase 2: per-row softmax pair, P = sm1 - lam*sm2 (bf16) ----
  for (int rr = 0; rr < 8; ++rr) {
    int m = w * 8 + rr;
    float v1[8], v2[8];
    float mx1 = -3.4e38f, mx2 = -3.4e38f;
#pragma unroll
    for (int q = 0; q < 8; ++q) {
      v1[q] = s1[m * SW + lane + 32 * q];
      v2[q] = s2[m * SW + lane + 32 * q];
      mx1 = fmaxf(mx1, v1[q]);
      mx2 = fmaxf(mx2, v2[q]);
    }
    for (int off = 16; off > 0; off >>= 1) {
      mx1 = fmaxf(mx1, __shfl_xor(mx1, off, 32));
      mx2 = fmaxf(mx2, __shfl_xor(mx2, off, 32));
    }
    float sum1 = 0.f, sum2 = 0.f;
#pragma unroll
    for (int q = 0; q < 8; ++q) {
      v1[q] = __expf(v1[q] - mx1); sum1 += v1[q];
      v2[q] = __expf(v2[q] - mx2); sum2 += v2[q];
    }
    for (int off = 16; off > 0; off >>= 1) {
      sum1 += __shfl_xor(sum1, off, 32);
      sum2 += __shfl_xor(sum2, off, 32);
    }
    float r1 = 1.f / sum1, r2 = lam / sum2;
#pragma unroll
    for (int q = 0; q < 8; ++q)
      sp[m * SP + lane + 32 * q] = (__bf16)(v1[q] * r1 - v2[q] * r2);
  }
  __syncthreads();

  // ---- phase 3: attn = P @ V  (A from LDS, B = VT rows) ----
  {
    const __bf16* Vb = VT + (size_t)b * 65536;
    int n0 = wn * 64;
    f32x8 acc[2][4];
#pragma unroll
    for (int i = 0; i < 2; ++i)
#pragma unroll
      for (int j = 0; j < 4; ++j) acc[i][j] = zero8();
    for (int k0 = 0; k0 < 256; k0 += 32) {
      bf16x16 a0 = load_frag(sp, SP, wm * 32,      k0, lane);
      bf16x16 a1 = load_frag(sp, SP, wm * 32 + 16, k0, lane);
#pragma unroll
      for (int j = 0; j < 4; ++j) {
        bf16x16 bf = load_frag(Vb, 256, n0 + 16 * j, k0, lane);
        acc[0][j] = wmma_bf16f32(a0, bf, acc[0][j]);
        acc[1][j] = wmma_bf16f32(a1, bf, acc[1][j]);
      }
    }
#pragma unroll
    for (int i = 0; i < 2; ++i)
#pragma unroll
      for (int j = 0; j < 4; ++j)
#pragma unroll
        for (int r = 0; r < 8; ++r) {
          int ml = wm * 32 + 16 * i + 8 * (lane >> 4) + r;
          int nl = n0 + 16 * j + (lane & 15);
          s1[ml * SW + nl] = acc[i][j][r];
        }
  }
  __syncthreads();

  // ---- phase 4a: row rmsnorm scale (over E=256) ----
  for (int rr = 0; rr < 8; ++rr) {
    int m = w * 8 + rr;
    float ss = 0.f;
#pragma unroll
    for (int q = 0; q < 8; ++q) {
      float v = s1[m * SW + lane + 32 * q];
      ss += v * v;
    }
    for (int off = 16; off > 0; off >>= 1) ss += __shfl_xor(ss, off, 32);
    if (lane == 0)
      rowsc[m] = (1.f - LAMBDA_INIT) / (sqrtf(ss) * 0.0625f + 1e-8f);
  }
  __syncthreads();

  // ---- phase 4b: x1[b][s][c] = x[b][s][c] + normScale[s]*attn[c][s]*rowsc[c]
  int cl = t & 63, nq = t >> 6;
  int cg = mb + cl;
  const float* xb = x + (size_t)b * 65536;
  float* ob = x1 + (size_t)b * 65536;
  float rs = rowsc[cl];
  for (int n = nq; n < 256; n += 4) {
    float v = s1[cl * SW + n] * rs * normScale[n] + xb[(size_t)n * 256 + cg];
    ob[(size_t)n * 256 + cg] = v;
  }
}

// ---------------------------------------------------------------------------
// K6: ln_ff over channel dim + transpose to bf16 hT[b][c][s]
// ---------------------------------------------------------------------------
__global__ __launch_bounds__(256) void da_prep_ffn_kernel(
    const float* __restrict__ x1, const float* __restrict__ lnff,
    __bf16* __restrict__ hT) {
  extern __shared__ float lds[];  // 64*257 + 64 floats
  const int SW = 257;
  float* rowinv = lds + 64 * SW;
  int t = threadIdx.x, b = blockIdx.x;
  int lane = t & 31, w = t >> 5;
  const float* xb = x1 + (size_t)b * 65536;

  for (int s0 = 0; s0 < 256; s0 += 64) {
    int c4 = (t & 63) * 4, sr = t >> 6;
    for (int r0 = 0; r0 < 64; r0 += 4) {
      int r = r0 + sr;
      F4f v = *reinterpret_cast<const F4f*>(xb + (size_t)(s0 + r) * 256 + c4);
      lds[r * SW + c4 + 0] = v.x; lds[r * SW + c4 + 1] = v.y;
      lds[r * SW + c4 + 2] = v.z; lds[r * SW + c4 + 3] = v.w;
    }
    __syncthreads();
    for (int rr = 0; rr < 8; ++rr) {
      int r = w * 8 + rr;
      float ss = 0.f;
#pragma unroll
      for (int q = 0; q < 8; ++q) {
        float v = lds[r * SW + lane + 32 * q];
        ss += v * v;
      }
      for (int off = 16; off > 0; off >>= 1) ss += __shfl_xor(ss, off, 32);
      if (lane == 0) rowinv[r] = 1.f / (sqrtf(ss) * 0.0625f + 1e-8f);
    }
    __syncthreads();
    int ds = t & 63, cq = t >> 6;
    float ri = rowinv[ds];
    for (int c = cq; c < 256; c += 4) {
      float v = lds[ds * SW + c] * ri * lnff[c];
      hT[(size_t)b * 65536 + (size_t)c * 256 + (s0 + ds)] = (__bf16)v;
    }
    __syncthreads();
  }
}

// ---------------------------------------------------------------------------
// Host side
// ---------------------------------------------------------------------------
static const size_t SZ_32M  = 33554432ull;
static const size_t OFF_LAM = 0;
static const size_t OFF_WQT = 1024;                  // 512x256 bf16
static const size_t OFF_WKT = OFF_WQT + 262144;
static const size_t OFF_WVT = OFF_WKT + 262144;      // 256x256 bf16
static const size_t OFF_W1T = OFF_WVT + 131072;      // 1024x256 bf16
static const size_t OFF_W2T = OFF_W1T + 524288;      // 256x1024 bf16
static const size_t OFF_AT  = 2ull * 1024 * 1024;
static const size_t OFF_TRT = OFF_AT + SZ_32M;
static const size_t OFF_SET = OFF_TRT + SZ_32M;
static const size_t OFF_Q   = OFF_SET + SZ_32M;      // 64MB
static const size_t OFF_K   = OFF_Q + 2 * SZ_32M;    // 64MB
static const size_t OFF_VT  = OFF_K + 2 * SZ_32M;    // 32MB
static const size_t OFF_X1  = OFF_VT + SZ_32M;       // 64MB f32
static const size_t OFF_HT  = OFF_TRT;               // reuse (trend dead)
static const size_t OFF_H1  = OFF_Q;                 // reuse (Q,K dead) 128MB

extern "C" void kernel_launch(void* const* d_in, const int* in_sizes, int n_in,
                              void* d_out, int out_size, void* d_ws, size_t ws_size,
                              hipStream_t stream) {
  (void)in_sizes; (void)n_in; (void)out_size; (void)ws_size;
  const float* x      = (const float*)d_in[0];
  const float* Wq     = (const float*)d_in[1];
  const float* bq     = (const float*)d_in[2];
  const float* Wk     = (const float*)d_in[3];
  const float* bk     = (const float*)d_in[4];
  const float* Wv     = (const float*)d_in[5];
  const float* bv     = (const float*)d_in[6];
  const float* lq1    = (const float*)d_in[7];
  const float* lk1    = (const float*)d_in[8];
  const float* lq2    = (const float*)d_in[9];
  const float* lk2    = (const float*)d_in[10];
  const float* lnattn = (const float*)d_in[11];
  const float* nscale = (const float*)d_in[12];
  const float* lnff   = (const float*)d_in[13];
  const float* W1     = (const float*)d_in[14];
  const float* b1     = (const float*)d_in[15];
  const float* W2     = (const float*)d_in[16];
  const float* b2     = (const float*)d_in[17];

  char* ws = (char*)d_ws;
  float*  lam = (float*)(ws + OFF_LAM);
  __bf16* WqT = (__bf16*)(ws + OFF_WQT);
  __bf16* WkT = (__bf16*)(ws + OFF_WKT);
  __bf16* WvT = (__bf16*)(ws + OFF_WVT);
  __bf16* W1T = (__bf16*)(ws + OFF_W1T);
  __bf16* W2T = (__bf16*)(ws + OFF_W2T);
  __bf16* aT  = (__bf16*)(ws + OFF_AT);
  __bf16* trT = (__bf16*)(ws + OFF_TRT);
  __bf16* seT = (__bf16*)(ws + OFF_SET);
  __bf16* Qb  = (__bf16*)(ws + OFF_Q);
  __bf16* Kb  = (__bf16*)(ws + OFF_K);
  __bf16* VTb = (__bf16*)(ws + OFF_VT);
  float*  x1  = (float*) (ws + OFF_X1);
  __bf16* hT  = (__bf16*)(ws + OFF_HT);
  __bf16* H1  = (__bf16*)(ws + OFF_H1);

  // setup: lambda + weight transposes (f32 -> bf16, [N][K] layout)
  da_lam_kernel<<<1, 256, 0, stream>>>(lq1, lk1, lq2, lk2, lam);
  da_tconv_kernel<<<512, 256, 0, stream>>>(Wq, WqT, 256, 512);
  da_tconv_kernel<<<512, 256, 0, stream>>>(Wk, WkT, 256, 512);
  da_tconv_kernel<<<256, 256, 0, stream>>>(Wv, WvT, 256, 256);
  da_tconv_kernel<<<1024, 256, 0, stream>>>(W1, W1T, 256, 1024);
  da_tconv_kernel<<<1024, 256, 0, stream>>>(W2, W2T, 1024, 256);

  // K1: normalize + decompose -> aT / trendT / seasonalT (bf16, transposed)
  da_prep_attn_kernel<<<256, 256, 256 * 257 * 4, stream>>>(x, lnattn, aT, trT, seT);

  // Projections (bf16 WMMA, f32 acc): Q/K row-major, V stored transposed
  da_gemm_kernel<0, 0><<<dim3(2, 4, 256), 256, 0, stream>>>(
      trT, 65536, 256, WqT, 0, 256, bq, Qb, 131072, 512, nullptr, 0, 256);
  da_gemm_kernel<0, 0><<<dim3(2, 4, 256), 256, 0, stream>>>(
      seT, 65536, 256, WkT, 0, 256, bk, Kb, 131072, 512, nullptr, 0, 256);
  da_gemm_kernel<1, 0><<<dim3(1, 4, 256), 256, 0, stream>>>(
      aT, 65536, 256, WvT, 0, 256, bv, VTb, 65536, 256, nullptr, 0, 256);

  // Fused differential attention + rmsnorm + residual
  size_t attn_smem = (size_t)(64 * 265 * 4) * 2 + 64 * 264 * 2 + 64 * 4;
  da_attn_kernel<<<dim3(4, 256), 256, attn_smem, stream>>>(
      Qb, Kb, VTb, lam, nscale, x, x1);

  // FFN: ln_ff + transpose, SiLU GEMM, final GEMM with transposed residual add
  da_prep_ffn_kernel<<<256, 256, (64 * 257 + 64) * 4, stream>>>(x1, lnff, hT);
  da_gemm_kernel<0, 1><<<dim3(4, 4, 256), 256, 0, stream>>>(
      hT, 65536, 256, W1T, 0, 256, b1, H1, 262144, 1024, nullptr, 0, 256);
  da_gemm_kernel<2, 0><<<dim3(1, 4, 256), 256, 0, stream>>>(
      H1, 262144, 1024, W2T, 0, 1024, b2, (float*)d_out, 65536, 256, x1, 65536, 1024);
}
